// Mamba_encoder_31421980738098
// MI455X (gfx1250) — compile-verified, matrix operand order checked by reading, not run on hardware
//
#include <hip/hip_runtime.h>
#include <hip/hip_bf16.h>

#define DIMD     512
#define D_STATE  16
#define D_CONV   4
#define D_INNER  1024
#define DT_RANK  32
#define BB       4
#define LLEN     4096
#define BLROWS   (BB * LLEN)   /* 16384 token rows */

typedef __bf16 bf16_t;
typedef __attribute__((ext_vector_type(16))) __bf16 v16bf;
typedef __attribute__((ext_vector_type(8)))  __bf16 v8bf;
typedef __attribute__((ext_vector_type(8)))  float  v8f;

__device__ __forceinline__ float sigmoidf_(float x) { return 1.0f / (1.0f + __expf(-x)); }
__device__ __forceinline__ float siluf_(float x)    { return x * sigmoidf_(x); }
__device__ __forceinline__ float softplusf_(float x){ return (x > 20.0f) ? x : log1pf(__expf(x)); }

__device__ __forceinline__ void split1(float f, bf16_t& h, bf16_t& l) {
  h = (bf16_t)f;
  l = (bf16_t)(f - (float)h);
}

// ---------------------------------------------------------------------------
// bf16 fragment load (pre-split matrices).  Per-lane WMMA layout (ISA 7.12.2,
// 16-bit 16x32 operand): row = lane&15, kb = k0 + (lane>>4)*8;
// halves 0..7 -> K = kb..kb+7, halves 8..15 -> K = kb+16..kb+23.
// Two 16-byte vector loads per fragment.
// ---------------------------------------------------------------------------
__device__ __forceinline__ v16bf load_frag_bf16(const bf16_t* __restrict__ base, int ld,
                                                int row, int kbase) {
  const bf16_t* p = base + (size_t)row * ld + kbase;
  v8bf a = *(const v8bf*)(p);
  v8bf b = *(const v8bf*)(p + 16);
  return __builtin_shufflevector(a, b, 0,1,2,3,4,5,6,7,8,9,10,11,12,13,14,15);
}

// fp32 fragment load with on-the-fly hi/lo split (used only for small-K GEMM3 A)
__device__ __forceinline__ void load_frag_split(const float* __restrict__ base, int ld,
                                                int row, int kbase,
                                                v16bf& hi, v16bf& lo) {
  const float* p = base + (size_t)row * ld + kbase;
#pragma unroll
  for (int i = 0; i < 8; ++i) {
    bf16_t h, l;
    split1(p[i], h, l);      hi[i] = h;     lo[i] = l;
    split1(p[16 + i], h, l); hi[8 + i] = h; lo[8 + i] = l;
  }
}

__device__ __forceinline__ v8f wmma_bf16(v16bf a, v16bf b, v8f c) {
  return __builtin_amdgcn_wmma_f32_16x16x32_bf16(false, a, false, b, (short)0, c, false, false);
}

// ---------------------------------------------------------------------------
// C[M x N] = act( A[M x K] @ W[N x K]^T + bias ), fp32 accumulate/out.
// A, W are pre-split bf16 hi/lo (or A fp32 on-the-fly when AFP32=1).
// Block = 128 threads = 4 waves; wave tile = 16 x 64 (4 accumulators,
// A fragment reused 4x; 12 WMMAs per k-step per wave).
// EPI: 0 = none, 1 = bias + softplus.
// ---------------------------------------------------------------------------
template <int EPI, int AFP32>
__global__ __launch_bounds__(128)
void wmma_gemm_kernel(const bf16_t* __restrict__ Ah, const bf16_t* __restrict__ Al,
                      const float* __restrict__ Af,
                      const bf16_t* __restrict__ Wh, const bf16_t* __restrict__ Wl,
                      const float* __restrict__ bias, float* __restrict__ C,
                      int M, int N, int K, int lda, int ldw, int ldc) {
  const int lane = threadIdx.x & 31;
  const int wave = threadIdx.x >> 5;
  const int m0 = blockIdx.x * 16;
  const int n0 = blockIdx.y * 256 + wave * 64;
  if (n0 >= N) return;                       // wave-uniform exit; EXEC stays all-1s

  const int rowA = m0 + (lane & 15);
  const int nW   = (lane & 15);
  const int koff = (lane >> 4) * 8;

  v8f acc[4] = {{}, {}, {}, {}};

  for (int k0 = 0; k0 < K; k0 += 32) {
    const int kb = k0 + koff;
    v16bf ah, al;
    if (AFP32) {
      load_frag_split(Af, lda, rowA, kb, ah, al);
    } else {
      ah = load_frag_bf16(Ah, lda, rowA, kb);
      al = load_frag_bf16(Al, lda, rowA, kb);
    }
#pragma unroll
    for (int t = 0; t < 4; ++t) {
      v16bf wh = load_frag_bf16(Wh, ldw, n0 + t * 16 + nW, kb);
      v16bf wl = load_frag_bf16(Wl, ldw, n0 + t * 16 + nW, kb);
      // split-precision product: hi*hi + hi*lo + lo*hi (lo*lo negligible)
      acc[t] = wmma_bf16(ah, wh, acc[t]);
      acc[t] = wmma_bf16(ah, wl, acc[t]);
      acc[t] = wmma_bf16(al, wh, acc[t]);
    }
  }

  const int mbase = m0 + (lane >> 4) * 8;
#pragma unroll
  for (int t = 0; t < 4; ++t) {
    const int col = n0 + t * 16 + (lane & 15);
    float bv = (EPI == 1) ? bias[col] : 0.0f;
#pragma unroll
    for (int r = 0; r < 8; ++r) {
      float v = acc[t][r];
      if (EPI == 1) v = softplusf_(v + bv);
      C[(size_t)(mbase + r) * ldc + col] = v;
    }
  }
}

// ---------------------------------------------------------------------------
// One-time fp32 -> bf16 hi/lo split (weights and x).
// ---------------------------------------------------------------------------
__global__ __launch_bounds__(256)
void split_bf16_kernel(const float* __restrict__ src, bf16_t* __restrict__ h,
                       bf16_t* __restrict__ l, int n) {
  int i = blockIdx.x * 256 + threadIdx.x;
  if (i >= n) return;
  bf16_t hh, ll;
  split1(src[i], hh, ll);
  h[i] = hh; l[i] = ll;
}

// ---------------------------------------------------------------------------
// Depthwise causal conv (width 4) + bias + silu.  xp lives in xz[:, 0:1024]
// (row stride 2048).  Emits xc fp32 (for the scan) plus bf16 hi/lo (for GEMM2).
// ---------------------------------------------------------------------------
__global__ __launch_bounds__(256)
void conv_silu_kernel(const float* __restrict__ xz, const float* __restrict__ Wc,
                      const float* __restrict__ bc, float* __restrict__ xc,
                      bf16_t* __restrict__ xch, bf16_t* __restrict__ xcl) {
  size_t idx = (size_t)blockIdx.x * 256 + threadIdx.x;
  if (idx >= (size_t)BLROWS * D_INNER) return;
  const int d = (int)(idx & (D_INNER - 1));
  const size_t bl = idx >> 10;
  const int l = (int)(bl & (LLEN - 1));
  float acc = bc[d];
#pragma unroll
  for (int j = 0; j < D_CONV; ++j) {
    const int dl = j - (D_CONV - 1);        // -3..0
    if (l + dl >= 0)
      acc += xz[(bl + dl) * (size_t)(2 * D_INNER) + d] * Wc[d * D_CONV + j];
  }
  const float s = siluf_(acc);
  xc[idx] = s;
  bf16_t h, lo;
  split1(s, h, lo);
  xch[idx] = h; xcl[idx] = lo;
}

// ---------------------------------------------------------------------------
// Selective scan.  Block = 256 threads = 16 channels x 16 states.
// Grid = B * (D_INNER/16) = 256 blocks.  Per-step operands staged in LDS in
// 64-step chunks; 16-lane shuffle reduction over the state dimension.
// Output ys written into the dead xp half of xz (row stride 2048).
// ---------------------------------------------------------------------------
__global__ __launch_bounds__(256)
void scan_kernel(const float* __restrict__ dt, const float* __restrict__ xc,
                 const float* __restrict__ xdbl, const float* __restrict__ A_log,
                 float* __restrict__ ys) {
  const int b    = blockIdx.x >> 6;
  const int dblk = blockIdx.x & 63;
  const int tid  = threadIdx.x;
  const int n    = tid & 15;
  const int dl   = tid >> 4;
  const int d    = dblk * 16 + dl;

  const float Adn = -__expf(A_log[d * D_STATE + n]);
  float h = 0.0f;

  __shared__ float sdt[64 * 16];
  __shared__ float sx [64 * 16];
  __shared__ float sB [64 * 16];
  __shared__ float sC [64 * 16];

  const size_t row0 = (size_t)b * LLEN;

  for (int l0 = 0; l0 < LLEN; l0 += 64) {
#pragma unroll
    for (int k = 0; k < 4; ++k) {
      const int idx = tid + k * 256;         // 0..1023
      const int s = idx >> 4;
      const int c = idx & 15;
      const size_t r = row0 + l0 + s;
      sdt[idx] = dt[r * D_INNER + dblk * 16 + c];
      sx [idx] = xc[r * D_INNER + dblk * 16 + c];
      sB [idx] = xdbl[r * 64 + DT_RANK + c];
      sC [idx] = xdbl[r * 64 + DT_RANK + D_STATE + c];
    }
    __syncthreads();

    for (int s = 0; s < 64; ++s) {
      const float dtv = sdt[s * 16 + dl];
      const float xv  = sx [s * 16 + dl];
      const float Bv  = sB [s * 16 + n];
      const float Cv  = sC [s * 16 + n];
      const float dA  = __expf(dtv * Adn);
      h = dA * h + dtv * xv * Bv;
      float contrib = h * Cv;
      contrib += __shfl_xor(contrib, 1, 16);
      contrib += __shfl_xor(contrib, 2, 16);
      contrib += __shfl_xor(contrib, 4, 16);
      contrib += __shfl_xor(contrib, 8, 16);
      if (n == 0)
        ys[(row0 + l0 + s) * (size_t)(2 * D_INNER) + d] = contrib;
    }
    __syncthreads();
  }
}

// ---------------------------------------------------------------------------
// y_act = (ys + xc * D) * silu(z); emitted directly as bf16 hi/lo pair
// (its only consumer is the WMMA output projection).
// ---------------------------------------------------------------------------
__global__ __launch_bounds__(256)
void combine_kernel(const float* __restrict__ xz, const float* __restrict__ xc,
                    const float* __restrict__ Dp,
                    bf16_t* __restrict__ yh, bf16_t* __restrict__ yl) {
  size_t idx = (size_t)blockIdx.x * 256 + threadIdx.x;
  if (idx >= (size_t)BLROWS * D_INNER) return;
  const int d = (int)(idx & (D_INNER - 1));
  const size_t bl = idx >> 10;
  const float ysv = xz[bl * (size_t)(2 * D_INNER) + d];
  const float zv  = xz[bl * (size_t)(2 * D_INNER) + D_INNER + d];
  const float v = (ysv + xc[idx] * Dp[d]) * siluf_(zv);
  bf16_t h, lo;
  split1(v, h, lo);
  yh[idx] = h; yl[idx] = lo;
}

// ---------------------------------------------------------------------------
// Row LayerNorm + residual + silu.  One block (256 threads) per token row of 512.
// ---------------------------------------------------------------------------
__global__ __launch_bounds__(256)
void ln_residual_kernel(const float* __restrict__ pre, const float* __restrict__ x,
                        const float* __restrict__ w, const float* __restrict__ b,
                        float* __restrict__ out) {
  const int row = blockIdx.x;
  const int t = threadIdx.x;
  const float* pr = pre + (size_t)row * DIMD;
  const float v0 = pr[t];
  const float v1 = pr[t + 256];

  __shared__ float sred[256];
  sred[t] = v0 + v1;
  __syncthreads();
  for (int o = 128; o > 0; o >>= 1) { if (t < o) sred[t] += sred[t + o]; __syncthreads(); }
  const float mu = sred[0] * (1.0f / DIMD);
  __syncthreads();

  const float d0 = v0 - mu, d1 = v1 - mu;
  sred[t] = d0 * d0 + d1 * d1;
  __syncthreads();
  for (int o = 128; o > 0; o >>= 1) { if (t < o) sred[t] += sred[t + o]; __syncthreads(); }
  const float rstd = rsqrtf(sred[0] * (1.0f / DIMD) + 1e-5f);

  const float o0 = d0 * rstd * w[t]       + b[t];
  const float o1 = d1 * rstd * w[t + 256] + b[t + 256];
  const float* xr = x + (size_t)row * DIMD;
  out[(size_t)row * DIMD + t]       = xr[t]       + siluf_(o0);
  out[(size_t)row * DIMD + t + 256] = xr[t + 256] + siluf_(o1);
}

// ---------------------------------------------------------------------------
extern "C" void kernel_launch(void* const* d_in, const int* in_sizes, int n_in,
                              void* d_out, int out_size, void* d_ws, size_t ws_size,
                              hipStream_t stream) {
  const float* x      = (const float*)d_in[0];   // (4,4096,512)
  const float* W_in   = (const float*)d_in[1];   // (2048,512)
  const float* W_conv = (const float*)d_in[2];   // (1024,1,4)
  const float* b_conv = (const float*)d_in[3];   // (1024,)
  const float* W_xprj = (const float*)d_in[4];   // (64,1024)
  const float* W_dt   = (const float*)d_in[5];   // (1024,32)
  const float* b_dt   = (const float*)d_in[6];   // (1024,)
  const float* A_log  = (const float*)d_in[7];   // (1024,16)
  const float* Dp     = (const float*)d_in[8];   // (1024,)
  const float* W_out  = (const float*)d_in[9];   // (512,1024)
  const float* ln_w   = (const float*)d_in[10];  // (512,)
  const float* ln_b   = (const float*)d_in[11];  // (512,)
  float* out = (float*)d_out;

  // ---- workspace layout (byte-based, 256B-aligned chunks) ----
  char* p = (char*)d_ws;
  auto take = [&](size_t bytes) { char* q = p; p += (bytes + 255) & ~(size_t)255; return q; };

  float*  xz     = (float*)take((size_t)BLROWS * 2 * D_INNER * 4);   // 134 MB
  float*  xc     = (float*)take((size_t)BLROWS * D_INNER * 4);       //  67 MB
  float*  dtbuf  = (float*)take((size_t)BLROWS * D_INNER * 4);       //  67 MB (outpre overlays after scan)
  float*  xdbl   = (float*)take((size_t)BLROWS * 64 * 4);            //   4 MB
  bf16_t* xh     = (bf16_t*)take((size_t)BLROWS * DIMD * 2);
  bf16_t* xl     = (bf16_t*)take((size_t)BLROWS * DIMD * 2);
  bf16_t* xch    = (bf16_t*)take((size_t)BLROWS * D_INNER * 2);
  bf16_t* xcl    = (bf16_t*)take((size_t)BLROWS * D_INNER * 2);
  bf16_t* yh     = (bf16_t*)take((size_t)BLROWS * D_INNER * 2);
  bf16_t* yl     = (bf16_t*)take((size_t)BLROWS * D_INNER * 2);
  bf16_t* winh   = (bf16_t*)take((size_t)2 * D_INNER * DIMD * 2);
  bf16_t* winl   = (bf16_t*)take((size_t)2 * D_INNER * DIMD * 2);
  bf16_t* wxh    = (bf16_t*)take((size_t)64 * D_INNER * 2);
  bf16_t* wxl    = (bf16_t*)take((size_t)64 * D_INNER * 2);
  bf16_t* wdh    = (bf16_t*)take((size_t)D_INNER * DT_RANK * 2);
  bf16_t* wdl    = (bf16_t*)take((size_t)D_INNER * DT_RANK * 2);
  bf16_t* woh    = (bf16_t*)take((size_t)DIMD * D_INNER * 2);
  bf16_t* wol    = (bf16_t*)take((size_t)DIMD * D_INNER * 2);
  float*  outpre = dtbuf;   // dt is dead after the scan

  const int MT = BLROWS / 16;   // 1024 M-tiles
  auto cdiv = [](int a, int b) { return (a + b - 1) / b; };

  // 0) one-time bf16 hi/lo splits
  { int n = BLROWS * DIMD;          split_bf16_kernel<<<cdiv(n,256),256,0,stream>>>(x,      xh,   xl,   n); }
  { int n = 2 * D_INNER * DIMD;     split_bf16_kernel<<<cdiv(n,256),256,0,stream>>>(W_in,   winh, winl, n); }
  { int n = 64 * D_INNER;           split_bf16_kernel<<<cdiv(n,256),256,0,stream>>>(W_xprj, wxh,  wxl,  n); }
  { int n = D_INNER * DT_RANK;      split_bf16_kernel<<<cdiv(n,256),256,0,stream>>>(W_dt,   wdh,  wdl,  n); }
  { int n = DIMD * D_INNER;         split_bf16_kernel<<<cdiv(n,256),256,0,stream>>>(W_out,  woh,  wol,  n); }

  // 1) xz = x @ W_in^T                       (16384 x 2048, K=512)
  wmma_gemm_kernel<0,0><<<dim3(MT, 2048/256), 128, 0, stream>>>(
      xh, xl, nullptr, winh, winl, nullptr, xz,
      BLROWS, 2048, DIMD, DIMD, DIMD, 2 * D_INNER);

  // 2) xc = silu(depthwise_conv(xp) + b_conv)  (+ bf16 split)
  conv_silu_kernel<<<(BLROWS * D_INNER) / 256, 256, 0, stream>>>(xz, W_conv, b_conv, xc, xch, xcl);

  // 3) xdbl = xc @ W_xproj^T                 (16384 x 64, K=1024)
  wmma_gemm_kernel<0,0><<<dim3(MT, 1), 128, 0, stream>>>(
      xch, xcl, nullptr, wxh, wxl, nullptr, xdbl,
      BLROWS, 64, D_INNER, D_INNER, D_INNER, 64);

  // 4) dt = softplus(dt_r @ W_dt^T + b_dt)   (16384 x 1024, K=32; A fp32 on-the-fly)
  wmma_gemm_kernel<1,1><<<dim3(MT, 1024/256), 128, 0, stream>>>(
      nullptr, nullptr, xdbl, wdh, wdl, b_dt, dtbuf,
      BLROWS, D_INNER, DT_RANK, 64, DT_RANK, D_INNER);

  // 5) selective scan -> ys (stored into xp half of xz)
  scan_kernel<<<BB * (D_INNER / 16), 256, 0, stream>>>(dtbuf, xc, xdbl, A_log, xz);

  // 6) y_act = (ys + xc*D) * silu(z) -> bf16 hi/lo
  combine_kernel<<<(BLROWS * D_INNER) / 256, 256, 0, stream>>>(xz, xc, Dp, yh, yl);

  // 7) outpre = y_act @ W_out^T              (16384 x 512, K=1024)
  wmma_gemm_kernel<0,0><<<dim3(MT, 2), 128, 0, stream>>>(
      yh, yl, nullptr, woh, wol, nullptr, outpre,
      BLROWS, DIMD, D_INNER, D_INNER, D_INNER, DIMD);

  // 8) out = x + silu(LN(outpre))
  ln_residual_kernel<<<BLROWS, 256, 0, stream>>>(outpre, x, ln_w, ln_b, out);
}
